// RandComb_model_8057358647983
// MI455X (gfx1250) — matrix operand order repaired and checked by reading, not compiled, not run on hardware
//
#include <hip/hip_runtime.h>

// ---------------------------------------------------------------------------
// RandComb model on MI455X (gfx1250, wave32, WMMA 16x16x32 f16, f32 accum).
// Random-channel gather folded into per-depth effective weights (W_eff);
// whole net = 7 dense WMMA GEMMs with implicit channels-last im2col.
// Round 3: NBLK register blocking -> 32x32 tiles (2 loads/WMMA) where N is
// an even tile count (conv3, convc, fc1, fc2); 32x16 elsewhere.
// ---------------------------------------------------------------------------

typedef __attribute__((ext_vector_type(16))) _Float16 v16h;
typedef __attribute__((ext_vector_type(8)))  _Float16 v8h;
typedef __attribute__((ext_vector_type(8)))  float    v8f;

#define DEV __device__ __forceinline__

DEV v8h h8_zero() {
    v8h z;
#pragma unroll
    for (int i = 0; i < 8; ++i) z[i] = (_Float16)0.0f;
    return z;
}
DEV v8h ld8(const _Float16* p) { return *(const v8h*)p; }
DEV v16h h16_combine(v8h lo, v8h hi) {
    v16h r;
#pragma unroll
    for (int i = 0; i < 8; ++i) { r[i] = lo[i]; r[i + 8] = hi[i]; }
    return r;
}
DEV void mma(v16h a, v16h b, v8f& acc) {
    acc = __builtin_amdgcn_wmma_f32_16x16x32_f16(false, a, false, b,
                                                 (short)0, acc, false, false);
}

enum { M_CONV1 = 0, M_CONV2 = 1, M_CONV3 = 2, M_CONVC = 3, M_FC_RELU = 4, M_FC_OUT = 5 };

// Workspace layout (units: _Float16 elements). All offsets 16B-aligned.
constexpr size_t SZ_CUBE = (size_t)256 * 49 * 160;
constexpr size_t SZ_Y1   = (size_t)138 * 256 * 49 * 8;
constexpr size_t SZ_Y2   = (size_t)134 * 256 * 49 * 16;
constexpr size_t SZ_Y3   = (size_t)256 * 25 * 4224;
constexpr size_t SZ_BT1  = (size_t)1104 * 1440;
constexpr size_t SZ_BT2  = (size_t)16 * 384;
constexpr size_t SZ_BT3  = (size_t)32 * 448;
constexpr size_t SZ_BTC  = (size_t)64 * 38016;
constexpr size_t SZ_FCIN = (size_t)256 * 576;
constexpr size_t SZ_BTF1 = (size_t)256 * 576;
constexpr size_t SZ_H1   = (size_t)256 * 256;
constexpr size_t SZ_BTF2 = (size_t)128 * 256;
constexpr size_t SZ_H2   = (size_t)256 * 128;
constexpr size_t SZ_BTF3 = (size_t)16 * 128;

constexpr size_t OFF_CUBE = 0;
constexpr size_t OFF_Y1   = OFF_CUBE + SZ_CUBE;
constexpr size_t OFF_Y2   = OFF_Y1 + SZ_Y1;
constexpr size_t OFF_Y3   = OFF_Y2 + SZ_Y2;
constexpr size_t OFF_BT1  = OFF_Y3 + SZ_Y3;
constexpr size_t OFF_BT2  = OFF_BT1 + SZ_BT1;
constexpr size_t OFF_BT3  = OFF_BT2 + SZ_BT2;
constexpr size_t OFF_BTC  = OFF_BT3 + SZ_BT3;
constexpr size_t OFF_FCIN = OFF_BTC + SZ_BTC;
constexpr size_t OFF_BTF1 = OFF_FCIN + SZ_FCIN;
constexpr size_t OFF_H1   = OFF_BTF1 + SZ_BTF1;
constexpr size_t OFF_BTF2 = OFF_H1 + SZ_H1;
constexpr size_t OFF_H2   = OFF_BTF2 + SZ_BTF2;
constexpr size_t OFF_BTF3 = OFF_H2 + SZ_H2;
constexpr size_t WS_HALVES = OFF_BTF3 + SZ_BTF3;

// ---- generic implicit-A fragment (CONV2 / CONV3 / FC; s = logical k start)
template <int MODE, int KPAD>
DEV v16h load_a(const _Float16* __restrict__ A, int d, int row, int s) {
    if constexpr (MODE == M_CONV2) {
        int n = row / 25, p = row % 25, ph = p / 5, pw = p % 5;
        int Ta = s >> 3;                           // 8 ch per tap -> 2 taps
        v8h lo = h8_zero(), hi = h8_zero();
        if (Ta < 45) {
            int kd = Ta / 9, t = Ta % 9, kh = t / 3, kw = t % 3;
            lo = ld8(A + ((size_t)((d + kd) * 256 + n) * 49 + (ph + kh) * 7 + (pw + kw)) * 8);
        }
        int Tb = Ta + 1;
        if (Tb < 45) {
            int kd = Tb / 9, t = Tb % 9, kh = t / 3, kw = t % 3;
            hi = ld8(A + ((size_t)((d + kd) * 256 + n) * 49 + (ph + kh) * 7 + (pw + kw)) * 8);
        }
        return h16_combine(lo, hi);
    } else if constexpr (MODE == M_CONV3) {
        int n = row / 25, p = row % 25, ph = p / 5, pw = p % 5;
        int T = s >> 4;                            // 16 ch per tap -> 1 tap
        if (T < 27) {
            int kd = T / 9, t = T % 9, kh = t / 3, kw = t % 3;
            const _Float16* ptr =
                A + ((size_t)((d + kd) * 256 + n) * 49 + (ph + kh) * 7 + (pw + kw)) * 16;
            return h16_combine(ld8(ptr), ld8(ptr + 8));
        }
        return h16_combine(h8_zero(), h8_zero());
    } else {                                       // dense FC: [M][KPAD]
        const _Float16* ptr = A + (size_t)row * KPAD + s;
        return h16_combine(ld8(ptr), ld8(ptr + 8));
    }
}

// ---- per-mode writeback with fused bias (+ReLU); Nstride = FC row stride
template <int MODE>
DEV void store_out(float v, int row, int col, int d, int Nstride,
                   const float* __restrict__ bias, _Float16* __restrict__ outH,
                   float* __restrict__ outF) {
    if constexpr (MODE == M_CONV1) {
        int dc = col >> 3, f = col & 7;            // col = d*8 + f
        int n = row / 25, p = row % 25;
        float r = fmaxf(v + bias[f], 0.0f);
        outH[((size_t)(dc * 256 + n) * 49 + (p / 5 + 1) * 7 + (p % 5 + 1)) * 8 + f] =
            (_Float16)r;
    } else if constexpr (MODE == M_CONV2) {
        int n = row / 25, p = row % 25;
        float r = fmaxf(v + bias[col], 0.0f);
        outH[((size_t)(d * 256 + n) * 49 + (p / 5 + 1) * 7 + (p % 5 + 1)) * 16 + col] =
            (_Float16)r;
    } else if constexpr (MODE == M_CONV3) {
        int n = row / 25, p = row % 25;
        float r = fmaxf(v + bias[col], 0.0f);
        outH[(size_t)(n * 25 + p) * 4224 + col * 132 + d] = (_Float16)r;
    } else if constexpr (MODE == M_CONVC) {
        int n = row / 9, q = row % 9;
        float r = fmaxf(v + bias[col], 0.0f);
        outH[(size_t)n * 576 + col * 9 + q] = (_Float16)r;
    } else if constexpr (MODE == M_FC_RELU) {
        float r = fmaxf(v + bias[col], 0.0f);
        outH[(size_t)row * Nstride + col] = (_Float16)r;
    } else {                                       // M_FC_OUT: f32 logits
        outF[(size_t)row * 16 + col] = v + bias[col];
    }
}

// ---- GEMM: one wave computes a 32x(16*NBLK) tile. B stored transposed
// [N][KPAD]; per lane lo=Bt[n][k0+16g..], hi=Bt[n][k0+16g+8..] (contiguous K
// order k=16g+s, same as A, so contiguous b128 loads work).
template <int MODE, int KPAD, int KCHUNKS, int NBLK>
__global__ __launch_bounds__(32) void gemm_wmma(
    const _Float16* __restrict__ A, const _Float16* __restrict__ Bt,
    const float* __restrict__ bias, _Float16* __restrict__ outH,
    float* __restrict__ outF, int Mtiles2, int Ntiles) {
    int lane = threadIdx.x & 31;
    int g = lane >> 4, ln = lane & 15;
    int blk = blockIdx.x;
    int nt = blk % Ntiles;
    int t2 = blk / Ntiles;
    int mt = t2 % Mtiles2;
    int d = t2 / Mtiles2;                          // depth for CONV2/CONV3
    int arow0 = mt * 32 + ln, arow1 = arow0 + 16;

    const _Float16* btRow[NBLK];
#pragma unroll
    for (int u = 0; u < NBLK; ++u)
        btRow[u] = Bt + (size_t)((nt * NBLK + u) * 16 + ln) * KPAD + 16 * g;

    v8f acc0[NBLK], acc1[NBLK];
#pragma unroll
    for (int u = 0; u < NBLK; ++u)
#pragma unroll
        for (int i = 0; i < 8; ++i) { acc0[u][i] = 0.0f; acc1[u][i] = 0.0f; }

    if constexpr (MODE == M_CONV1) {
        // taps (kh,kw) x 5 chunks of 32 ch; cube_cl [n][7][7][160]
        int n0 = arow0 / 25, p0 = arow0 % 25;
        int n1 = arow1 / 25, p1 = arow1 % 25;
        int rb0 = (n0 * 49 + (p0 / 5) * 7 + (p0 % 5)) * 160 + 16 * g;
        int rb1 = (n1 * 49 + (p1 / 5) * 7 + (p1 % 5)) * 160 + 16 * g;
#pragma unroll
        for (int T = 0; T < 9; ++T) {
            const _Float16* pa0 = A + rb0 + ((T / 3) * 7 + (T % 3)) * 160;
            const _Float16* pa1 = A + rb1 + ((T / 3) * 7 + (T % 3)) * 160;
#pragma unroll
            for (int j = 0; j < 5; ++j) {
                v16h a0 = h16_combine(ld8(pa0 + j * 32), ld8(pa0 + j * 32 + 8));
                v16h a1 = h16_combine(ld8(pa1 + j * 32), ld8(pa1 + j * 32 + 8));
#pragma unroll
                for (int u = 0; u < NBLK; ++u) {
                    const _Float16* pb = btRow[u] + T * 160 + j * 32;
                    v16h b = h16_combine(ld8(pb), ld8(pb + 8));
                    mma(a0, b, acc0[u]);
                    mma(a1, b, acc1[u]);
                }
            }
        }
    } else if constexpr (MODE == M_CONVC) {
        // taps (kh,kw) x 132 chunks of 32 ch; y3_cl [n][5][5][4224]
        int n0 = arow0 / 9, q0 = arow0 % 9;
        int n1 = arow1 / 9, q1 = arow1 % 9;
        int rb0 = (n0 * 25 + (q0 / 3) * 5 + (q0 % 3)) * 4224 + 16 * g;
        int rb1 = (n1 * 25 + (q1 / 3) * 5 + (q1 % 3)) * 4224 + 16 * g;
#pragma unroll
        for (int T = 0; T < 9; ++T) {
            const _Float16* pa0 = A + rb0 + ((T / 3) * 5 + (T % 3)) * 4224;
            const _Float16* pa1 = A + rb1 + ((T / 3) * 5 + (T % 3)) * 4224;
#pragma unroll 2
            for (int j = 0; j < 132; ++j) {
                v16h a0 = h16_combine(ld8(pa0 + j * 32), ld8(pa0 + j * 32 + 8));
                v16h a1 = h16_combine(ld8(pa1 + j * 32), ld8(pa1 + j * 32 + 8));
#pragma unroll
                for (int u = 0; u < NBLK; ++u) {
                    const _Float16* pb = btRow[u] + T * 4224 + j * 32;
                    v16h b = h16_combine(ld8(pb), ld8(pb + 8));
                    mma(a0, b, acc0[u]);
                    mma(a1, b, acc1[u]);
                }
            }
        }
    } else {
#pragma unroll
        for (int ks = 0; ks < KCHUNKS; ++ks) {
            int k0 = ks * 32;
            v16h a0 = load_a<MODE, KPAD>(A, d, arow0, k0 + 16 * g);
            v16h a1 = load_a<MODE, KPAD>(A, d, arow1, k0 + 16 * g);
#pragma unroll
            for (int u = 0; u < NBLK; ++u) {
                v16h b = h16_combine(ld8(btRow[u] + k0), ld8(btRow[u] + k0 + 8));
                mma(a0, b, acc0[u]);
                mma(a1, b, acc1[u]);
            }
        }
    }

    int Nstride = Ntiles * NBLK * 16;              // FC row stride
#pragma unroll
    for (int u = 0; u < NBLK; ++u) {
        int col = (nt * NBLK + u) * 16 + ln;
#pragma unroll
        for (int i = 0; i < 8; ++i) {              // C/D: lanes16-31 hold M=8..15
            int row = mt * 32 + 8 * g + i;
            store_out<MODE>(acc0[u][i], row, col, d, Nstride, bias, outH, outF);
            store_out<MODE>(acc1[u][i], row + 16, col, d, Nstride, bias, outH, outF);
        }
    }
}

// ------------------------------ prep kernels -------------------------------
__global__ void k_zero8(_Float16* p, long n8) {
    long i = (long)blockIdx.x * blockDim.x + threadIdx.x;
    if (i < n8) ((v8h*)p)[i] = h8_zero();
}

// crop x[:, :, 10:15, 10:15] -> channels-last padded cube [n][7][7][160]
__global__ void k_cube(const float* __restrict__ x, _Float16* __restrict__ cube) {
    int idx = blockIdx.x * 256 + threadIdx.x;
    if (idx >= 256 * 144 * 25) return;
    int n = idx / (144 * 25), r = idx % (144 * 25);
    int c = r / 25, p = r % 25, ph = p / 5, pw = p % 5;
    float v = x[((size_t)(n * 144 + c) * 25 + (10 + ph)) * 25 + (10 + pw)];
    cube[((size_t)(n * 49 + (ph + 1) * 7 + (pw + 1))) * 160 + c] = (_Float16)v;
}

// per-depth effective conv1 weights: Bt1[d*8+f][T*160+c] (f16, transposed)
__global__ __launch_bounds__(256) void k_weff(const float* __restrict__ w1,
                                              const int* __restrict__ rl,
                                              _Float16* __restrict__ Bt1) {
    __shared__ float s[8 * 9 * 144];               // 41.5 KB
    int d = blockIdx.x, tid = threadIdx.x;
    for (int i = tid; i < 8 * 9 * 144; i += 256) s[i] = 0.0f;
    __syncthreads();
    for (int idx = tid; idx < 700 * 72; idx += 256) {
        int pair = idx / 72, e = idx % 72;
        int i = pair / 7, kd = pair % 7;
        int c = rl[i * 144 + d + kd];
        int f = e / 9, t = e % 9;
        atomicAdd(&s[(f * 9 + t) * 144 + c],
                  w1[((size_t)(f * 100 + i) * 7 + kd) * 9 + t]);
    }
    __syncthreads();
    for (int idx = tid; idx < 8 * 1440; idx += 256) {
        int f = idx / 1440, k = idx % 1440;
        int T = k / 160, c = k % 160;
        float v = (c < 144) ? s[(f * 9 + T) * 144 + c] : 0.0f;
        Bt1[(size_t)(d * 8 + f) * 1440 + k] = (_Float16)v;
    }
}

__global__ void k_bt2(const float* __restrict__ w2, _Float16* __restrict__ Bt) {
    int idx = blockIdx.x * 256 + threadIdx.x;
    if (idx >= 16 * 384) return;
    int f = idx / 384, k = idx % 384, T = k / 8, ci = k % 8;
    float v = 0.0f;
    if (T < 45) { int kd = T / 9, t = T % 9; v = w2[((size_t)(f * 8 + ci) * 5 + kd) * 9 + t]; }
    Bt[idx] = (_Float16)v;
}
__global__ void k_bt3(const float* __restrict__ w3, _Float16* __restrict__ Bt) {
    int idx = blockIdx.x * 256 + threadIdx.x;
    if (idx >= 32 * 448) return;
    int f = idx / 448, k = idx % 448, T = k / 16, ci = k % 16;
    float v = 0.0f;
    if (T < 27) { int kd = T / 9, t = T % 9; v = w3[((size_t)(f * 16 + ci) * 3 + kd) * 9 + t]; }
    Bt[idx] = (_Float16)v;
}
__global__ void k_btc(const float* __restrict__ wc, _Float16* __restrict__ Bt) {
    int idx = blockIdx.x * 256 + threadIdx.x;
    if (idx >= 64 * 38016) return;
    int o = idx / 38016, k = idx % 38016, T = k / 4224, ch = k % 4224;
    Bt[idx] = (_Float16)wc[((size_t)o * 4224 + ch) * 9 + T];
}
// dense FC weight transpose+convert: src f32 [K][N] -> dst f16 [N][K]
__global__ void k_tr(const float* __restrict__ src, _Float16* __restrict__ dst,
                     int K, int N) {
    int idx = blockIdx.x * 256 + threadIdx.x;
    if (idx >= K * N) return;
    int nn = idx / K, kk = idx % K;
    dst[idx] = (_Float16)src[(size_t)kk * N + nn];
}

// ------------------------------- launcher ----------------------------------
extern "C" void kernel_launch(void* const* d_in, const int* in_sizes, int n_in,
                              void* d_out, int out_size, void* d_ws, size_t ws_size,
                              hipStream_t stream) {
    const float* x   = (const float*)d_in[0];
    const int*   rl  = (const int*)d_in[1];
    const float* w1  = (const float*)d_in[2];
    const float* b1  = (const float*)d_in[3];
    const float* w2  = (const float*)d_in[4];
    const float* b2  = (const float*)d_in[5];
    const float* w3  = (const float*)d_in[6];
    const float* b3  = (const float*)d_in[7];
    const float* wc  = (const float*)d_in[8];
    const float* bc  = (const float*)d_in[9];
    const float* fw1 = (const float*)d_in[10];
    const float* fb1 = (const float*)d_in[11];
    const float* fw2 = (const float*)d_in[12];
    const float* fb2 = (const float*)d_in[13];
    const float* fw3 = (const float*)d_in[14];
    const float* fb3 = (const float*)d_in[15];
    float* out = (float*)d_out;

    if (ws_size < WS_HALVES * sizeof(_Float16)) return;  // ~149 MB needed
    _Float16* ws   = (_Float16*)d_ws;
    _Float16* cube = ws + OFF_CUBE;
    _Float16* y1   = ws + OFF_Y1;
    _Float16* y2   = ws + OFF_Y2;
    _Float16* y3   = ws + OFF_Y3;
    _Float16* bt1  = ws + OFF_BT1;
    _Float16* bt2  = ws + OFF_BT2;
    _Float16* bt3  = ws + OFF_BT3;
    _Float16* btc  = ws + OFF_BTC;
    _Float16* fcin = ws + OFF_FCIN;
    _Float16* btf1 = ws + OFF_BTF1;
    _Float16* h1   = ws + OFF_H1;
    _Float16* btf2 = ws + OFF_BTF2;
    _Float16* h2   = ws + OFF_H2;
    _Float16* btf3 = ws + OFF_BTF3;

    // zero padded feature maps (cube, y1, y2 contiguous at ws start)
    long zn8 = (long)((SZ_CUBE + SZ_Y1 + SZ_Y2) / 8);
    k_zero8<<<(unsigned)((zn8 + 255) / 256), 256, 0, stream>>>(cube, zn8);

    k_cube<<<(256 * 144 * 25 + 255) / 256, 256, 0, stream>>>(x, cube);
    k_weff<<<138, 256, 0, stream>>>(w1, rl, bt1);
    k_bt2<<<(16 * 384 + 255) / 256, 256, 0, stream>>>(w2, bt2);
    k_bt3<<<(32 * 448 + 255) / 256, 256, 0, stream>>>(w3, bt3);
    k_btc<<<(64 * 38016 + 255) / 256, 256, 0, stream>>>(wc, btc);
    k_tr<<<(576 * 256 + 255) / 256, 256, 0, stream>>>(fw1, btf1, 576, 256);
    k_tr<<<(256 * 128 + 255) / 256, 256, 0, stream>>>(fw2, btf2, 256, 128);
    k_tr<<<(128 * 16 + 255) / 256, 256, 0, stream>>>(fw3, btf3, 128, 16);

    // conv1 (gather folded into W_eff): 6400 x 1440 x 1104 (N=69 tiles, NBLK=1)
    gemm_wmma<M_CONV1, 1440, 45, 1><<<200 * 69, 32, 0, stream>>>(
        cube, bt1, b1, y1, nullptr, 200, 69);
    // conv2: 134 depths x (6400 x 384 x 16), NBLK=1
    gemm_wmma<M_CONV2, 384, 12, 1><<<134 * 200, 32, 0, stream>>>(
        y1, bt2, b2, y2, nullptr, 200, 1);
    // conv3: 132 depths x (6400 x 448 x 32), 32x32 tiles (NBLK=2)
    gemm_wmma<M_CONV3, 448, 14, 2><<<132 * 200, 32, 0, stream>>>(
        y2, bt3, b3, y3, nullptr, 200, 1);
    // convc: 2304 x 38016 x 64, 32x32 tiles
    gemm_wmma<M_CONVC, 38016, 1188, 2><<<72 * 2, 32, 0, stream>>>(
        y3, btc, bc, fcin, nullptr, 72, 2);
    // fc1: 256 x 576 x 256 (relu), 32x32 tiles
    gemm_wmma<M_FC_RELU, 576, 18, 2><<<8 * 8, 32, 0, stream>>>(
        fcin, btf1, fb1, h1, nullptr, 8, 8);
    // fc2: 256 x 256 x 128 (relu), 32x32 tiles
    gemm_wmma<M_FC_RELU, 256, 8, 2><<<8 * 4, 32, 0, stream>>>(
        h1, btf2, fb2, h2, nullptr, 8, 4);
    // fc3: 256 x 128 x 16 -> f32 logits
    gemm_wmma<M_FC_OUT, 128, 4, 1><<<8, 32, 0, stream>>>(
        h2, btf3, fb3, nullptr, out, 8, 1);
}
